// OpenaiTransformer_33131377722065
// MI455X (gfx1250) — matrix operand order repaired and checked by reading, not compiled, not run on hardware
//
#include <hip/hip_runtime.h>
#include <math.h>

// ---------------- problem constants ----------------
#define BB   16
#define SS   512
#define DD   768
#define HH   12
#define LL   12
#define DH   64
#define FF3  3072
#define D3   2304
#define NTOK (BB * SS)           // 8192 rows

typedef __attribute__((ext_vector_type(16))) __bf16 bf16x16;
typedef __attribute__((ext_vector_type(8)))  __bf16 bf16x8;
typedef __attribute__((ext_vector_type(4)))  __bf16 bf16x4;
typedef __attribute__((ext_vector_type(8)))  float  v8f;

__device__ __forceinline__ v8f wmma_bf16(bf16x16 a, bf16x16 b, v8f c) {
  return __builtin_amdgcn_wmma_f32_16x16x32_bf16(false, a, false, b, (short)0, c, false, false);
}

__device__ __forceinline__ bf16x16 join16(bf16x8 lo, bf16x8 hi) {
  return __builtin_shufflevector(lo, hi, 0, 1, 2, 3, 4, 5, 6, 7, 8, 9, 10, 11, 12, 13, 14, 15);
}

__device__ __forceinline__ bf16x4 cvt4(float4 v) {
  bf16x4 o = {(__bf16)v.x, (__bf16)v.y, (__bf16)v.z, (__bf16)v.w};
  return o;
}

__device__ __forceinline__ float gelu_f(float x) {
  const float k = 0.7978845608028654f; // sqrt(2/pi)
  return 0.5f * x * (1.0f + tanhf(k * (x + 0.044715f * x * x * x)));
}

// ---------------- embedding: h0 = we[x0] + we[x1] ----------------
__global__ __launch_bounds__(256) void embed_kernel(const int* __restrict__ x,
                                                    const float* __restrict__ we,
                                                    float* __restrict__ h0) {
  const int row = blockIdx.x;
  const int i0 = x[row * 2 + 0];
  const int i1 = x[row * 2 + 1];
  const float* w0 = we + (size_t)i0 * DD;
  const float* w1 = we + (size_t)i1 * DD;
  float* o = h0 + (size_t)row * DD;
  for (int c = threadIdx.x; c < DD; c += 256) o[c] = w0[c] + w1[c];
}

// ---------------- WMMA GEMM: Out[M,N] = A[M,K] @ Bw[K,N] + bias (+res) (gelu) ----------------
// epi: 0 = bias, 1 = bias + residual, 2 = bias + gelu
#define BM 128
#define BN 128
#define BK 64
#define LP 72     // LDS row pitch (bf16 elements), 144B: multiple of 16B

__global__ __launch_bounds__(256) void gemm_wmma(const float* __restrict__ A,
                                                 const float* __restrict__ Bw,
                                                 const float* __restrict__ bias,
                                                 const float* __restrict__ Res,
                                                 float* __restrict__ Out,
                                                 int M, int N, int K, int epi) {
  __shared__ __bf16 As[2][BM][LP];   // [m][k]
  __shared__ __bf16 Bt[2][BN][LP];   // transposed: [n][k]

  const int tid  = threadIdx.x;
  const int lane = tid & 31;
  const int wave = tid >> 5;          // 0..7
  const int wm   = wave & 1;          // 2 waves over M (64 rows each)
  const int wn   = wave >> 1;         // 4 waves over N (32 cols each)
  const int bm   = blockIdx.y * BM;
  const int bn   = blockIdx.x * BN;

  const int ln  = lane & 15;
  const int hi  = lane >> 4;
  const int kbA = hi * 8;
  const int kbB = hi * 8;

  v8f acc[4][2] = {};

  float4 aR[8];    // register staging for A tile (128x64 f32)
  float4 bR[8];    // register staging for B tile (64x128 f32)

  const int nT = K / BK;

  // ---- tile load helpers (inlined by macros of code) ----
  // A: 128x64 -> 2048 float4, 8 per thread. idx: r = idx>>4, c4 = (idx&15)*4
  // B: 4x4 sub-blocks, 512 blocks, 2 per thread: kb=(blk>>5)*4, nb=(blk&31)*4
#define LOAD_TILE(t)                                                              \
  {                                                                               \
    const int k0 = (t) * BK;                                                      \
    _Pragma("unroll") for (int i = 0; i < 8; ++i) {                               \
      int idx = tid + i * 256;                                                    \
      int r = idx >> 4, c4 = (idx & 15) * 4;                                      \
      aR[i] = *reinterpret_cast<const float4*>(&A[(size_t)(bm + r) * K + k0 + c4]); \
    }                                                                             \
    _Pragma("unroll") for (int j = 0; j < 2; ++j) {                               \
      int blk = tid + j * 256;                                                    \
      int kb = (blk >> 5) * 4, nb = (blk & 31) * 4;                               \
      _Pragma("unroll") for (int rr = 0; rr < 4; ++rr)                            \
        bR[j * 4 + rr] = *reinterpret_cast<const float4*>(                        \
            &Bw[(size_t)(k0 + kb + rr) * N + bn + nb]);                           \
    }                                                                             \
  }

#define STORE_TILE(buf)                                                           \
  {                                                                               \
    _Pragma("unroll") for (int i = 0; i < 8; ++i) {                               \
      int idx = tid + i * 256;                                                    \
      int r = idx >> 4, c4 = (idx & 15) * 4;                                      \
      *reinterpret_cast<bf16x4*>(&As[buf][r][c4]) = cvt4(aR[i]);                  \
    }                                                                             \
    _Pragma("unroll") for (int j = 0; j < 2; ++j) {                               \
      int blk = tid + j * 256;                                                    \
      int kb = (blk >> 5) * 4, nb = (blk & 31) * 4;                               \
      _Pragma("unroll") for (int cc = 0; cc < 4; ++cc) {                          \
        bf16x4 col = {(__bf16)reinterpret_cast<const float*>(&bR[j * 4 + 0])[cc], \
                      (__bf16)reinterpret_cast<const float*>(&bR[j * 4 + 1])[cc], \
                      (__bf16)reinterpret_cast<const float*>(&bR[j * 4 + 2])[cc], \
                      (__bf16)reinterpret_cast<const float*>(&bR[j * 4 + 3])[cc]};\
        *reinterpret_cast<bf16x4*>(&Bt[buf][nb + cc][kb]) = col;                  \
      }                                                                           \
    }                                                                             \
  }

  LOAD_TILE(0);
  STORE_TILE(0);

  for (int t = 0; t < nT; ++t) {
    const int cur = t & 1;
    __syncthreads();
    if (t + 1 < nT) LOAD_TILE(t + 1);

#pragma unroll
    for (int ks = 0; ks < 2; ++ks) {
      bf16x16 bfr[2];
#pragma unroll
      for (int nt = 0; nt < 2; ++nt) {
        const int col = wn * 32 + nt * 16 + ln;
        bf16x8 lo = *reinterpret_cast<const bf16x8*>(&Bt[cur][col][ks * 32 + kbB]);
        bf16x8 hv = *reinterpret_cast<const bf16x8*>(&Bt[cur][col][ks * 32 + kbB + 16]);
        bfr[nt] = join16(lo, hv);
      }
#pragma unroll
      for (int mt = 0; mt < 4; ++mt) {
        const int row = wm * 64 + mt * 16 + ln;
        bf16x8 lo = *reinterpret_cast<const bf16x8*>(&As[cur][row][ks * 32 + kbA]);
        bf16x8 hv = *reinterpret_cast<const bf16x8*>(&As[cur][row][ks * 32 + kbA + 16]);
        bf16x16 af = join16(lo, hv);
        acc[mt][0] = wmma_bf16(af, bfr[0], acc[mt][0]);
        acc[mt][1] = wmma_bf16(af, bfr[1], acc[mt][1]);
      }
    }

    if (t + 1 < nT) STORE_TILE((t + 1) & 1);
  }
#undef LOAD_TILE
#undef STORE_TILE

  // epilogue (C layout: lane ln = col, VGPR r -> row r + hi*8)
#pragma unroll
  for (int mt = 0; mt < 4; ++mt) {
#pragma unroll
    for (int nt = 0; nt < 2; ++nt) {
      const int gn = bn + wn * 32 + nt * 16 + ln;
      const float bv = bias[gn];
#pragma unroll
      for (int r = 0; r < 8; ++r) {
        const int gm = bm + wm * 64 + mt * 16 + r + hi * 8;
        float v = acc[mt][nt][r] + bv;
        if (epi == 1) v += Res[(size_t)gm * N + gn];
        if (epi == 2) v = gelu_f(v);
        Out[(size_t)gm * N + gn] = v;
      }
    }
  }
}

// ---------------- fused causal attention per (b, h, 64-row query tile) ----------------
// qkv: [NTOK, 2304]  (q | k | v, head h at col h*64)
// aout: [NTOK, 768]
__global__ __launch_bounds__(256) void attn_kernel(const float* __restrict__ qkv,
                                                   float* __restrict__ aout) {
  const int qt = blockIdx.x;     // 0..7 (64 q rows each)
  const int h  = blockIdx.y;
  const int b  = blockIdx.z;

  __shared__ __bf16 p[64][SS + 8];                // probs (bf16), pitch 520 (1040B, 16B-mult)
  __shared__ __bf16 qs[64][LP];
  __shared__ float  lsum[64];
  __shared__ __align__(16) __bf16 kvbuf[128 * LP]; // K tile [key][d] OR V^T tile [d][key]
  __bf16 (*kk)[LP]   = reinterpret_cast<__bf16(*)[LP]>(kvbuf);   // 128 x 72
  __bf16 (*vts)[136] = reinterpret_cast<__bf16(*)[136]>(kvbuf);  // 64 x 136 (fits)

  const int tid  = threadIdx.x;
  const int lane = tid & 31;
  const int wave = tid >> 5;
  const int ln   = lane & 15;
  const int hi   = lane >> 4;
  const int kbA  = hi * 8;
  const int kbB  = hi * 8;

  const size_t rowbase = (size_t)b * SS;
  const int nk = (qt >> 1) + 1;                   // 128-key tiles covering causal span

  // phase 0: load Q tile (64x64)
#pragma unroll
  for (int i = 0; i < 4; ++i) {
    int idx = tid + i * 256;
    int r = idx >> 4, c4 = (idx & 15) * 4;
    const float4 v = *reinterpret_cast<const float4*>(
        &qkv[(rowbase + qt * 64 + r) * D3 + h * DH + c4]);
    *reinterpret_cast<bf16x4*>(&qs[r][c4]) = cvt4(v);
  }
  __syncthreads();

  const int mw  = (wave & 3) * 16;        // 4 waves over 64 q rows
  const int nwS = (wave >> 2) * 4;        // phase1: 2 wave groups over 8 key n-tiles

  // hoisted Q fragments (two 32-wide k-steps over DH=64)
  bf16x16 aq[2];
#pragma unroll
  for (int ks = 0; ks < 2; ++ks) {
    bf16x8 lo = *reinterpret_cast<const bf16x8*>(&qs[mw + ln][ks * 32 + kbA]);
    bf16x8 hv = *reinterpret_cast<const bf16x8*>(&qs[mw + ln][ks * 32 + kbA + 16]);
    aq[ks] = join16(lo, hv);
  }

  // phase 1: scores = (Q K^T) * 1/8 with causal mask -> p (bf16)
  for (int kt = 0; kt < nk; ++kt) {
    __syncthreads();
#pragma unroll
    for (int i = 0; i < 8; ++i) {          // K tile 128x64, row-major [key][d]
      int idx = tid + i * 256;
      int r = idx >> 4, c4 = (idx & 15) * 4;
      const float4 v = *reinterpret_cast<const float4*>(
          &qkv[(rowbase + kt * 128 + r) * D3 + DD + h * DH + c4]);
      *reinterpret_cast<bf16x4*>(&kk[r][c4]) = cvt4(v);
    }
    __syncthreads();
#pragma unroll
    for (int t = 0; t < 4; ++t) {
      const int nt = nwS + t;
      v8f c = {};
#pragma unroll
      for (int ks = 0; ks < 2; ++ks) {
        bf16x8 lo = *reinterpret_cast<const bf16x8*>(&kk[nt * 16 + ln][ks * 32 + kbB]);
        bf16x8 hv = *reinterpret_cast<const bf16x8*>(&kk[nt * 16 + ln][ks * 32 + kbB + 16]);
        c = wmma_bf16(aq[ks], join16(lo, hv), c);
      }
      const int key = kt * 128 + nt * 16 + ln;
#pragma unroll
      for (int r = 0; r < 8; ++r) {
        const int qr = mw + r + hi * 8;
        const int qg = qt * 64 + qr;
        float s = c[r] * 0.125f;           // 1/sqrt(64)
        if (key > qg) s = -1e30f;
        p[qr][key] = (__bf16)s;
      }
    }
  }
  __syncthreads();

  // phase 2: per-row softmax (unnormalized exp stored; row sum kept)
  if (tid < 64) {
    const int r = tid;
    const int ncol = nk * 128;
    float mx = -1e30f;
    for (int c = 0; c < ncol; ++c) mx = fmaxf(mx, (float)p[r][c]);
    float s = 0.0f;
    for (int c = 0; c < ncol; ++c) {
      float e = expf((float)p[r][c] - mx);
      s += e;
      p[r][c] = (__bf16)e;
    }
    lsum[r] = s;
  }

  // phase 3: O = P @ V (V staged transposed: vts[d][key])
  const int nwO = (wave >> 2) * 2;         // 2 wave groups over 4 d n-tiles
  v8f o[2] = {};
  for (int kt = 0; kt < nk; ++kt) {
    __syncthreads();
#pragma unroll
    for (int j = 0; j < 2; ++j) {          // V tile 128x64 in 4x4 blocks, transposed store
      int blk = tid + j * 256;
      int keyb = (blk & 31) * 4, db = (blk >> 5) * 4;
      float4 vr[4];
#pragma unroll
      for (int rr = 0; rr < 4; ++rr)
        vr[rr] = *reinterpret_cast<const float4*>(
            &qkv[(rowbase + kt * 128 + keyb + rr) * D3 + 2 * DD + h * DH + db]);
#pragma unroll
      for (int cc = 0; cc < 4; ++cc) {
        bf16x4 col = {(__bf16)reinterpret_cast<const float*>(&vr[0])[cc],
                      (__bf16)reinterpret_cast<const float*>(&vr[1])[cc],
                      (__bf16)reinterpret_cast<const float*>(&vr[2])[cc],
                      (__bf16)reinterpret_cast<const float*>(&vr[3])[cc]};
        *reinterpret_cast<bf16x4*>(&vts[db + cc][keyb]) = col;
      }
    }
    __syncthreads();
#pragma unroll
    for (int ks = 0; ks < 4; ++ks) {
      bf16x8 alo = *reinterpret_cast<const bf16x8*>(&p[mw + ln][kt * 128 + ks * 32 + kbA]);
      bf16x8 ahi = *reinterpret_cast<const bf16x8*>(&p[mw + ln][kt * 128 + ks * 32 + kbA + 16]);
      bf16x16 ap = join16(alo, ahi);
#pragma unroll
      for (int t = 0; t < 2; ++t) {
        const int dcol = (nwO + t) * 16 + ln;
        bf16x8 lo = *reinterpret_cast<const bf16x8*>(&vts[dcol][ks * 32 + kbB]);
        bf16x8 hv = *reinterpret_cast<const bf16x8*>(&vts[dcol][ks * 32 + kbB + 16]);
        o[t] = wmma_bf16(ap, join16(lo, hv), o[t]);
      }
    }
  }

  // phase 4: normalize by row sums, write to [NTOK, 768]
#pragma unroll
  for (int t = 0; t < 2; ++t) {
    const int d = h * DH + (nwO + t) * 16 + ln;
#pragma unroll
    for (int r = 0; r < 8; ++r) {
      const int qr = mw + r + hi * 8;
      aout[(rowbase + qt * 64 + qr) * (size_t)DD + d] = o[t][r] / lsum[qr];
    }
  }
}

// ---------------- LayerNorm over D=768 (one block per row) ----------------
__global__ __launch_bounds__(256) void ln_kernel(const float* __restrict__ X,
                                                 const float* __restrict__ g,
                                                 const float* __restrict__ bta,
                                                 float* __restrict__ Out) {
  __shared__ float red[256];
  const int row = blockIdx.x;
  const int tid = threadIdx.x;
  const float* xr = X + (size_t)row * DD;
  const float v0 = xr[tid], v1 = xr[tid + 256], v2 = xr[tid + 512];

  red[tid] = v0 + v1 + v2;
  __syncthreads();
  for (int off = 128; off > 0; off >>= 1) {
    if (tid < off) red[tid] += red[tid + off];
    __syncthreads();
  }
  const float mu = red[0] * (1.0f / DD);
  __syncthreads();

  const float d0 = v0 - mu, d1 = v1 - mu, d2 = v2 - mu;
  red[tid] = d0 * d0 + d1 * d1 + d2 * d2;
  __syncthreads();
  for (int off = 128; off > 0; off >>= 1) {
    if (tid < off) red[tid] += red[tid + off];
    __syncthreads();
  }
  const float rs = rsqrtf(red[0] * (1.0f / DD) + 1e-5f);

  float* o = Out + (size_t)row * DD;
  o[tid]       = d0 * rs * g[tid]       + bta[tid];
  o[tid + 256] = d1 * rs * g[tid + 256] + bta[tid + 256];
  o[tid + 512] = d2 * rs * g[tid + 512] + bta[tid + 512];
}

// ---------------- orchestration ----------------
extern "C" void kernel_launch(void* const* d_in, const int* in_sizes, int n_in,
                              void* d_out, int out_size, void* d_ws, size_t ws_size,
                              hipStream_t stream) {
  const int*   x    = (const int*)d_in[0];
  const float* we   = (const float*)d_in[1];
  const float* caw  = (const float*)d_in[2];
  const float* cab  = (const float*)d_in[3];
  const float* apw  = (const float*)d_in[4];
  const float* apb  = (const float*)d_in[5];
  const float* g1   = (const float*)d_in[6];
  const float* b1   = (const float*)d_in[7];
  const float* fw   = (const float*)d_in[8];
  const float* fb   = (const float*)d_in[9];
  const float* pw   = (const float*)d_in[10];
  const float* pb   = (const float*)d_in[11];
  const float* g2   = (const float*)d_in[12];
  const float* b2   = (const float*)d_in[13];

  float* out = (float*)d_out;                       // [13, NTOK, 768]
  const size_t slice = (size_t)NTOK * DD;

  // workspace layout (floats)
  float* f_buf  = (float*)d_ws;                     // NTOK*3072 (qkv then fc act)
  float* a_buf  = f_buf  + (size_t)NTOK * FF3;      // NTOK*768  (attn out, then res2)
  float* r1_buf = a_buf  + slice;                   // NTOK*768
  float* n_buf  = r1_buf + slice;                   // NTOK*768

  const dim3 blk(256);

  embed_kernel<<<NTOK, blk, 0, stream>>>(x, we, out);

  for (int l = 0; l < LL; ++l) {
    const float* hprev = out + (size_t)l * slice;
    float*       hout  = out + (size_t)(l + 1) * slice;

    const float* caw_l = caw + (size_t)l * DD * D3;
    const float* cab_l = cab + (size_t)l * D3;
    const float* apw_l = apw + (size_t)l * DD * DD;
    const float* apb_l = apb + (size_t)l * DD;
    const float* fw_l  = fw  + (size_t)l * DD * FF3;
    const float* fb_l  = fb  + (size_t)l * FF3;
    const float* pw_l  = pw  + (size_t)l * FF3 * DD;
    const float* pb_l  = pb  + (size_t)l * DD;

    gemm_wmma<<<dim3(D3 / BN, NTOK / BM), blk, 0, stream>>>(
        hprev, caw_l, cab_l, nullptr, f_buf, NTOK, D3, DD, 0);

    attn_kernel<<<dim3(SS / 64, HH, BB), blk, 0, stream>>>(f_buf, a_buf);

    gemm_wmma<<<dim3(DD / BN, NTOK / BM), blk, 0, stream>>>(
        a_buf, apw_l, apb_l, hprev, r1_buf, NTOK, DD, DD, 1);

    ln_kernel<<<NTOK, blk, 0, stream>>>(r1_buf, g1 + (size_t)l * DD, b1 + (size_t)l * DD, n_buf);

    gemm_wmma<<<dim3(FF3 / BN, NTOK / BM), blk, 0, stream>>>(
        n_buf, fw_l, fb_l, nullptr, f_buf, NTOK, FF3, DD, 2);

    gemm_wmma<<<dim3(DD / BN, NTOK / BM), blk, 0, stream>>>(
        f_buf, pw_l, pb_l, n_buf, a_buf, NTOK, DD, FF3, 1);

    ln_kernel<<<NTOK, blk, 0, stream>>>(a_buf, g2 + (size_t)l * DD, b2 + (size_t)l * DD, hout);
  }
}
